// DebugModel_14379550507707
// MI455X (gfx1250) — matrix operand order repaired and checked by reading, not compiled, not run on hardware
//
#include <hip/hip_runtime.h>
#include <math.h>

typedef __bf16 v16bf __attribute__((ext_vector_type(16)));
typedef float  v8f   __attribute__((ext_vector_type(8)));

#define WMMA_BF16(a, b, c) \
  __builtin_amdgcn_wmma_f32_16x16x32_bf16(false, (a), false, (b), (short)0, (c), false, false)

// ---------------------------------------------------------------------------
// CDNA5 async global->LDS copy (ASYNCcnt protocol, CDNA5 ISA)
// GV mode: VDST = LDS byte offset (VGPR), VADDR = 64-bit global address (VGPR pair)
// ---------------------------------------------------------------------------
__device__ __forceinline__ void async_load_b128(void* lds_ptr, const void* gptr) {
  uint32_t lds_off = (uint32_t)(size_t)lds_ptr;  // low 32 bits of generic addr = LDS offset
  asm volatile("global_load_async_to_lds_b128 %0, %1, off"
               :: "v"(lds_off), "v"((uint64_t)(size_t)gptr)
               : "memory");
}
__device__ __forceinline__ void wait_asynccnt0() {
  asm volatile("s_wait_asynccnt 0x0" ::: "memory");
}

// ---------------------------------------------------------------------------
// Fragment loaders use contiguous k order (k = 16*(lane/16) + s) for both A and B
// A (16x32 bf16, row-major in LDS): lanes 0-15 rows 0-15 / K 0..15,
// lanes 16-31 same rows / K 16..31  -> contiguous 32B per half-lane
// ---------------------------------------------------------------------------
__device__ __forceinline__ v16bf load_a_frag(const __bf16* base, int ld, int lane) {
  int r = lane & 15, kb = (lane >> 4) * 16;
  v16bf f;
#pragma unroll
  for (int j = 0; j < 8; ++j) {
    f[j]     = base[r * ld + kb + j];
    f[8 + j] = base[r * ld + kb + 8 + j];
  }
  return f;
}

// B operand where LDS holds the transpose (N-major): B[k][n] = base[n*ld + k]
// -> 16 contiguous halves per lane (2 x ds_load_b128)
__device__ __forceinline__ v16bf load_b_fragT(const __bf16* base, int ld, int lane) {
  int c = lane & 15, kb = (lane >> 4) * 16;
  v16bf f;
#pragma unroll
  for (int j = 0; j < 16; ++j) f[j] = base[c * ld + kb + j];
  return f;
}

// Same, but fp32 source (state matrix), converted on the fly
__device__ __forceinline__ v16bf load_b_fragT_f32(const float* base, int ld, int lane) {
  int c = lane & 15, kb = (lane >> 4) * 16;
  v16bf f;
#pragma unroll
  for (int j = 0; j < 16; ++j) f[j] = (__bf16)base[c * ld + kb + j];
  return f;
}

// ---------------------------------------------------------------------------
// fp32 -> bf16 conversion
// ---------------------------------------------------------------------------
__global__ void cvt_f32_bf16(const float* __restrict__ in, __bf16* __restrict__ out, int n) {
  int i = blockIdx.x * blockDim.x + threadIdx.x;
  if (i < n) out[i] = (__bf16)in[i];
}

// ---------------------------------------------------------------------------
// Generic bf16 GEMM: C[M,N] = A[M,K] @ B[K,N], fp32 out.
// 64x64 block tile, 256 threads (8 waves), 2 WMMA tiles / wave, K-step 32.
// A tile staged via async global->LDS DMA; B tile staged transposed.
// ---------------------------------------------------------------------------
__global__ __launch_bounds__(256) void gemm_bf16_f32(
    const __bf16* __restrict__ A, const __bf16* __restrict__ B,
    float* __restrict__ C, int M, int N, int K) {
  __shared__ __bf16 As[64][32];    // 64 rows x 32 K (row stride 64B, 16B aligned)
  __shared__ __bf16 BsT[64][40];   // N-major transpose, +8 pad (row stride 80B, 16B aligned)
  int tid = threadIdx.x;
  int lane = tid & 31, w = tid >> 5;
  int bm = blockIdx.y * 64, bn = blockIdx.x * 64;
  int tn = w & 3, tm0 = w >> 2, tm1 = tm0 + 2;

  v8f c0 = {};
  v8f c1 = {};
  for (int k0 = 0; k0 < K; k0 += 32) {
    // A tile: 4096B = 256 threads x 16B, contiguous both sides -> async DMA
    {
      int o = tid * 16;
      int row = o >> 6, cb = o & 63;
      async_load_b128((char*)&As[0][0] + o,
                      (const char*)A + ((size_t)(bm + row) * K + k0) * 2 + cb);
    }
    // B tile: coalesced global reads, transposed LDS writes
#pragma unroll
    for (int i = 0; i < 8; ++i) {
      int e = tid + i * 256;
      int kr = e >> 6, n = e & 63;
      BsT[n][kr] = B[(size_t)(k0 + kr) * N + bn + n];
    }
    wait_asynccnt0();
    __syncthreads();
    v16bf bfrag = load_b_fragT(&BsT[tn * 16][0], 40, lane);
    v16bf a0 = load_a_frag(&As[tm0 * 16][0], 32, lane);
    v16bf a1 = load_a_frag(&As[tm1 * 16][0], 32, lane);
    c0 = WMMA_BF16(a0, bfrag, c0);
    c1 = WMMA_BF16(a1, bfrag, c1);
    __syncthreads();
  }
  int hi = lane >> 4, col = lane & 15;
#pragma unroll
  for (int r = 0; r < 8; ++r) {
    int m0 = bm + tm0 * 16 + r + 8 * hi;
    int m1 = bm + tm1 * 16 + r + 8 * hi;
    C[(size_t)m0 * N + bn + tn * 16 + col] = c0[r];
    C[(size_t)m1 * N + bn + tn * 16 + col] = c1[r];
  }
}

// ---------------------------------------------------------------------------
// Low-rank forget gate: gk = log_sigmoid((x@Wgk1)@Wgk2 + b)/16, fp32 [4096,256]
// ---------------------------------------------------------------------------
__global__ __launch_bounds__(256) void lowrank_gk_kernel(
    const float* __restrict__ x, const float* __restrict__ W1,
    const float* __restrict__ W2, const float* __restrict__ b2,
    float* __restrict__ gkf) {
  __shared__ float t1[16];
  int row = blockIdx.x, tid = threadIdx.x;
  if (tid < 16) {
    float s = 0.f;
    for (int k = 0; k < 512; ++k) s += x[(size_t)row * 512 + k] * W1[k * 16 + tid];
    t1[tid] = s;
  }
  __syncthreads();
  float pre = b2[tid];
#pragma unroll
  for (int r = 0; r < 16; ++r) pre += t1[r] * W2[r * 256 + tid];
  float ls = fminf(pre, 0.f) - log1pf(__expf(-fabsf(pre)));  // stable log_sigmoid
  gkf[(size_t)row * 256 + tid] = ls * (1.f / 16.f);
}

// ---------------------------------------------------------------------------
// Chunked GLA scan. One block per (b,h) chain; 8 waves; chunk C=32.
// dk=64 per head, dv=128 per head. State kept dv-major (ST[dv][dk]) fp32 in LDS
// so its WMMA B-fragments are contiguous.
// ---------------------------------------------------------------------------
__global__ __launch_bounds__(256) void scan_kernel(
    const float* __restrict__ qf, const float* __restrict__ kf,
    const float* __restrict__ vf, const float* __restrict__ gkf,
    float* __restrict__ of, int T_) {
  __shared__ float  ST[128][64];    // 32KB fp32 state, transposed: ST[dv][dk]
  __shared__ float  gbuf[32][64];   // gate chunk -> in-place cumulative G_t[d]
  __shared__ float  gC[64];         // exp(G_C) per dk dim
  __shared__ __bf16 qs[32][64];     // q~ = q * exp(G_t) * scale      (A operand)
  __shared__ __bf16 ktN[32][64];    // k~ = k * exp(-G_t), time-major (B^T operand)
  __shared__ __bf16 khT[64][32];    // (k * exp(G_C - G_t))^T         (A operand)
  __shared__ __bf16 vsT[128][32];   // v chunk, dv-major              (B^T operand)
  __shared__ __bf16 ps[32][32];     // masked intra-chunk attention   (A operand)

  int tid = threadIdx.x, lane = tid & 31, w = tid >> 5;
  int b = blockIdx.x >> 2, h = blockIdx.x & 3;
  const float scale = 0.125f;  // dk=64 -> 1/8

  for (int i = tid; i < 128 * 64; i += 256) (&ST[0][0])[i] = 0.f;
  __syncthreads();

  for (int ci = 0; ci < T_ / 32; ++ci) {
    int row0 = b * T_ + ci * 32;

    // stage gate chunk: 8KB, rows contiguous (256B) both sides -> async DMA
#pragma unroll
    for (int i = 0; i < 2; ++i) {
      int o = (tid + i * 256) * 16;
      int s = o >> 8, cb = o & 255;
      async_load_b128((char*)&gbuf[0][0] + o,
                      (const char*)gkf + ((size_t)(row0 + s) * 256 + h * 64) * 4 + cb);
    }
    wait_asynccnt0();
    __syncthreads();
    if (tid < 64) {
      float run = 0.f;
#pragma unroll
      for (int s = 0; s < 32; ++s) { run += gbuf[s][tid]; gbuf[s][tid] = run; }
      gC[tid] = __expf(run);
    }
    __syncthreads();

    // stage scaled q/k (gates <=0, /16-normalized -> exponents bounded)
#pragma unroll
    for (int i = 0; i < 8; ++i) {
      int e = tid + i * 256, s = e >> 6, d = e & 63;
      float G = gbuf[s][d];
      float gTot = gbuf[31][d];
      size_t gi = (size_t)(row0 + s) * 256 + h * 64 + d;
      float qv = qf[gi], kv = kf[gi];
      qs[s][d]  = (__bf16)(qv * __expf(G) * scale);
      ktN[s][d] = (__bf16)(kv * __expf(-G));
      khT[d][s] = (__bf16)(kv * __expf(gTot - G));
    }
#pragma unroll
    for (int i = 0; i < 16; ++i) {
      int e = tid + i * 256, s = e >> 7, j = e & 127;
      vsT[j][s] = (__bf16)vf[(size_t)(row0 + s) * 512 + h * 128 + j];
    }
    __syncthreads();

    // ---- O_inter = q~ @ S_old : 2x8 tiles, wave w owns dv-tile w ----
    v8f o0 = {};
    v8f o1 = {};
    {
      int tn = w;
#pragma unroll
      for (int kk = 0; kk < 64; kk += 32) {
        v16bf bfrag = load_b_fragT_f32(&ST[tn * 16][kk], 64, lane);
        v16bf a0 = load_a_frag(&qs[0][kk], 64, lane);
        v16bf a1 = load_a_frag(&qs[16][kk], 64, lane);
        o0 = WMMA_BF16(a0, bfrag, o0);
        o1 = WMMA_BF16(a1, bfrag, o1);
      }
    }

    // ---- P = tril(q~ @ k~^T) : 2x2 tiles on waves 0..3 ----
    if (w < 4) {
      int tm = w >> 1, tn = w & 1;
      v8f p = {};
#pragma unroll
      for (int kk = 0; kk < 64; kk += 32) {
        v16bf a  = load_a_frag(&qs[tm * 16][kk], 64, lane);
        v16bf bb = load_b_fragT(&ktN[tn * 16][kk], 64, lane);
        p = WMMA_BF16(a, bb, p);
      }
      int hi = lane >> 4, col = lane & 15;
#pragma unroll
      for (int r = 0; r < 8; ++r) {
        int mm = tm * 16 + r + 8 * hi, nn = tn * 16 + col;
        ps[mm][nn] = (nn <= mm) ? (__bf16)p[r] : (__bf16)0.f;
      }
    }
    __syncthreads();  // ps ready; all ST reads (O_inter) complete

    // ---- O += P @ V ; write O chunk ----
    {
      int tn = w;
      v16bf b0 = load_b_fragT(&vsT[tn * 16][0], 32, lane);
      v16bf a0 = load_a_frag(&ps[0][0], 32, lane);
      v16bf a1 = load_a_frag(&ps[16][0], 32, lane);
      o0 = WMMA_BF16(a0, b0, o0);
      o1 = WMMA_BF16(a1, b0, o1);
      int hi = lane >> 4, col = lane & 15;
#pragma unroll
      for (int r = 0; r < 8; ++r) {
        int m0 = r + 8 * hi, m1 = 16 + r + 8 * hi;
        of[(size_t)(row0 + m0) * 512 + h * 128 + tn * 16 + col] = o0[r];
        of[(size_t)(row0 + m1) * 512 + h * 128 + tn * 16 + col] = o1[r];
      }
    }

    // ---- S = S * exp(G_C) + k^hat^T @ V : 4x8 tiles, 4 per wave ----
#pragma unroll
    for (int q4 = 0; q4 < 4; ++q4) {
      int ut = w + q4 * 8;
      int tm = ut >> 3, tn = ut & 7;
      v16bf a  = load_a_frag(&khT[tm * 16][0], 32, lane);
      v16bf bb = load_b_fragT(&vsT[tn * 16][0], 32, lane);
      v8f u = {};
      u = WMMA_BF16(a, bb, u);
      int hi = lane >> 4, col = lane & 15;
#pragma unroll
      for (int r = 0; r < 8; ++r) {
        int dkr = tm * 16 + r + 8 * hi, cc = tn * 16 + col;
        ST[cc][dkr] = ST[cc][dkr] * gC[dkr] + u[r];
      }
    }
    __syncthreads();
  }
}

// ---------------------------------------------------------------------------
// Fused per-head RMSNorm + swish gate; emits bf16 for the Wo GEMM.
// ---------------------------------------------------------------------------
__global__ __launch_bounds__(128) void normgate_kernel(
    const float* __restrict__ of, const float* __restrict__ gf,
    const float* __restrict__ gnw, __bf16* __restrict__ ogb) {
  __shared__ float red[128];
  int t = blockIdx.x >> 2, h = blockIdx.x & 3, j = threadIdx.x;
  size_t idx = (size_t)t * 512 + h * 128 + j;
  float v = of[idx];
  red[j] = v * v;
  __syncthreads();
  for (int s = 64; s > 0; s >>= 1) {
    if (j < s) red[j] += red[j + s];
    __syncthreads();
  }
  float rinv = rsqrtf(red[0] * (1.f / 128.f) + 1e-5f);
  float g = gf[idx];
  float sw = g / (1.f + __expf(-g));  // g * sigmoid(g)
  ogb[idx] = (__bf16)(v * rinv * gnw[j] * sw);
}

// ---------------------------------------------------------------------------
// Head projection: out[4096,10] = y1 @ Whead + bhead (tiny; scalar fp32)
// ---------------------------------------------------------------------------
__global__ void head_kernel(const float* __restrict__ y1, const float* __restrict__ Wh,
                            const float* __restrict__ bh, float* __restrict__ out, int M) {
  int i = blockIdx.x * blockDim.x + threadIdx.x;
  if (i >= M * 10) return;
  int r = i / 10, c = i % 10;
  float s = bh[c];
  for (int k = 0; k < 512; ++k) s += y1[(size_t)r * 512 + k] * Wh[k * 10 + c];
  out[i] = s;
}

// ---------------------------------------------------------------------------
extern "C" void kernel_launch(void* const* d_in, const int* in_sizes, int n_in,
                              void* d_out, int out_size, void* d_ws, size_t ws_size,
                              hipStream_t stream) {
  const float* x    = (const float*)d_in[0];
  const float* Wq   = (const float*)d_in[1];
  const float* Wk   = (const float*)d_in[2];
  const float* Wv   = (const float*)d_in[3];
  const float* Wg   = (const float*)d_in[4];
  const float* Wgk1 = (const float*)d_in[5];
  const float* Wgk2 = (const float*)d_in[6];
  const float* bgk2 = (const float*)d_in[7];
  const float* gnw  = (const float*)d_in[8];
  const float* Wo   = (const float*)d_in[9];
  const float* Wh   = (const float*)d_in[10];
  const float* bh   = (const float*)d_in[11];
  float* out = (float*)d_out;

  const int M = 2 * 2048;  // B*T tokens
  const int D = 512, DK = 256, DV = 512, T = 2048;

  char* ws = (char*)d_ws;
  size_t off = 0;
  auto alloc = [&](size_t bytes) -> void* {
    void* p = ws + off;
    off += (bytes + 255) & ~(size_t)255;
    return p;
  };
  __bf16* xb  = (__bf16*)alloc((size_t)M * D * 2);
  __bf16* Wqb = (__bf16*)alloc((size_t)D * DK * 2);
  __bf16* Wkb = (__bf16*)alloc((size_t)D * DK * 2);
  __bf16* Wvb = (__bf16*)alloc((size_t)D * DV * 2);
  __bf16* Wgb = (__bf16*)alloc((size_t)D * DV * 2);
  __bf16* Wob = (__bf16*)alloc((size_t)DV * D * 2);
  float*  qf  = (float*)alloc((size_t)M * DK * 4);
  float*  kf  = (float*)alloc((size_t)M * DK * 4);
  float*  vf  = (float*)alloc((size_t)M * DV * 4);
  float*  gf  = (float*)alloc((size_t)M * DV * 4);
  float*  gkf = (float*)alloc((size_t)M * DK * 4);
  float*  of  = (float*)alloc((size_t)M * DV * 4);
  __bf16* ogb = (__bf16*)alloc((size_t)M * DV * 2);
  float*  y1  = (float*)alloc((size_t)M * D * 4);

  auto cvt = [&](const float* src, __bf16* dst, int n) {
    cvt_f32_bf16<<<(n + 255) / 256, 256, 0, stream>>>(src, dst, n);
  };
  cvt(x, xb, M * D);
  cvt(Wq, Wqb, D * DK);
  cvt(Wk, Wkb, D * DK);
  cvt(Wv, Wvb, D * DV);
  cvt(Wg, Wgb, D * DV);
  cvt(Wo, Wob, DV * D);

  lowrank_gk_kernel<<<M, 256, 0, stream>>>(x, Wgk1, Wgk2, bgk2, gkf);

  gemm_bf16_f32<<<dim3(DK / 64, M / 64), 256, 0, stream>>>(xb, Wqb, qf, M, DK, D);
  gemm_bf16_f32<<<dim3(DK / 64, M / 64), 256, 0, stream>>>(xb, Wkb, kf, M, DK, D);
  gemm_bf16_f32<<<dim3(DV / 64, M / 64), 256, 0, stream>>>(xb, Wvb, vf, M, DV, D);
  gemm_bf16_f32<<<dim3(DV / 64, M / 64), 256, 0, stream>>>(xb, Wgb, gf, M, DV, D);

  scan_kernel<<<8, 256, 0, stream>>>(qf, kf, vf, gkf, of, T);

  normgate_kernel<<<M * 4, 128, 0, stream>>>(of, gf, gnw, ogb);

  gemm_bf16_f32<<<dim3(D / 64, M / 64), 256, 0, stream>>>(ogb, Wob, y1, M, D, DV);

  head_kernel<<<(M * 10 + 255) / 256, 256, 0, stream>>>(y1, Wh, bh, out, M);
}
